// Graph_Attention_Union_47957604827160
// MI455X (gfx1250) — compile-verified, matrix operand order checked by reading, not run on hardware
//
#include <hip/hip_runtime.h>
#include <hip/hip_bf16.h>

// ---------------------------------------------------------------------------
// Graph Attention Union for MI455X (gfx1250, wave32, WMMA).
// B=64, C=256, Hz=Wz=16 (Nz=256), Hx=Wx=32 (Nx=1024), O=256, H=8, dh=32.
// All GEMMs on v_wmma_f32_16x16x32_bf16 (f32 accum); staging fully vectorized
// (b128 global loads / ds stores) so the WMMA pipe is not issue-starved.
// ---------------------------------------------------------------------------

typedef unsigned short u16;
typedef __attribute__((ext_vector_type(16))) __bf16 v16bf;
typedef __attribute__((ext_vector_type(8)))  __bf16 v8bf;
typedef __attribute__((ext_vector_type(8)))  float  v8f;

union Frag  { v16bf v; v8bf h[2]; };
union Pack8 { v8bf  v; u16  u[8]; };

__device__ __forceinline__ u16 f2bf(float f) {
  unsigned u = __float_as_uint(f);
  u += 0x7FFFu + ((u >> 16) & 1u);           // round-to-nearest-even
  return (u16)(u >> 16);
}

// Load a 16x32 bf16 WMMA fragment (A: row=M, or B: row=N) from a row-major
// [row][k] buffer. Per ISA 7.12.2: lanes 0-15 hold row=lane, K {0..7,16..23};
// lanes 16-31 hold row=lane-16, K {8..15,24..31}. Two b128 loads per lane.
__device__ __forceinline__ v16bf ld_frag(const u16* base, int stride, int k0) {
  const int lane = threadIdx.x & 31;
  const u16* p = base + (size_t)(lane & 15) * stride + k0 + ((lane & 16) ? 8 : 0);
  Frag f;
  f.h[0] = *(const v8bf*)(p);
  f.h[1] = *(const v8bf*)(p + 16);
  return f.v;
}

#define WMMA_BF16(A, B, C) \
  __builtin_amdgcn_wmma_f32_16x16x32_bf16(false, (A), false, (B), (short)0, (C), false, false)

// ---------------------------------------------------------------------------
// Prep: convert weights to bf16 (stack [query;g] and [support;g]); fold BN.
// ---------------------------------------------------------------------------
__global__ void prep_kernel(const float* __restrict__ qw, const float* __restrict__ sw,
                            const float* __restrict__ gw, const float* __restrict__ fw,
                            const float* __restrict__ g_b, const float* __restrict__ g_gamma,
                            const float* __restrict__ g_beta, const float* __restrict__ g_mean,
                            const float* __restrict__ g_var,
                            const float* __restrict__ fi_b, const float* __restrict__ fi_gamma,
                            const float* __restrict__ fi_beta, const float* __restrict__ fi_mean,
                            const float* __restrict__ fi_var,
                            u16* __restrict__ wqg, u16* __restrict__ wsg, u16* __restrict__ wfi,
                            float* __restrict__ gs, float* __restrict__ gb2,
                            float* __restrict__ fs, float* __restrict__ fb2) {
  const int NW = 256 * 256;
  const int NF = 256 * 512;
  int i = blockIdx.x * blockDim.x + threadIdx.x;
  if (i < NW) {
    wqg[i] = f2bf(qw[i]);
    wsg[i] = f2bf(sw[i]);
  } else if (i < 2 * NW) {
    int j = i - NW;
    u16 v = f2bf(gw[j]);
    wqg[i] = v;                 // rows 256..511 of the stacked matrices
    wsg[i] = v;
  } else if (i < 2 * NW + NF) {
    int j = i - 2 * NW;
    wfi[j] = f2bf(fw[j]);
  } else if (i < 2 * NW + NF + 256) {
    int c = i - (2 * NW + NF);
    float inv = g_gamma[c] * rsqrtf(g_var[c] + 1e-5f);
    gs[c] = inv;
    gb2[c] = g_b[c] * inv + g_beta[c] - g_mean[c] * inv;
  } else if (i < 2 * NW + NF + 512) {
    int c = i - (2 * NW + NF + 256);
    float inv = fi_gamma[c] * rsqrtf(fi_var[c] + 1e-5f);
    fs[c] = inv;
    fb2[c] = fi_b[c] * inv + fi_beta[c] - fi_mean[c] * inv;
  }
}

// ---------------------------------------------------------------------------
// Projection GEMM: Y[512, NP] = Wstk[512,256] @ X[b][256][NP].
// Rows 0..255 -> q/k (+bias), rows 256..511 -> relu(bn(g)) routed to v or u.
// Block: 256 threads = 8 waves, tile 128(M) x 64(N), K chunked by 128.
// ---------------------------------------------------------------------------
#define PKC 128

__device__ __forceinline__ void proj_store(v8f acc, int ob, int p, int b, int NP, int is_zf,
                                           const float* __restrict__ qbias,
                                           const float* __restrict__ gscale,
                                           const float* __restrict__ gbias,
                                           u16* __restrict__ qk_out,
                                           u16* __restrict__ g_out) {
  // ob = first of 8 consecutive output channels held by this lane (ob % 8 == 0)
  Pack8 pk;
  if (ob < 256) {                                  // q or k path (+bias)
    int hh = ob >> 5, d0 = ob & 31;
#pragma unroll
    for (int r = 0; r < 8; ++r) pk.u[r] = f2bf(acc[r] + qbias[ob + r]);
    *(v8bf*)&qk_out[(((size_t)b * 8 + hh) * NP + p) * 32 + d0] = pk.v;
  } else {                                         // g path: relu(bn(.))
    int oc0 = ob - 256;
    if (is_zf) {
      // v layout [b][h][d][nz] (B-fragment-friendly for emb GEMM) - scattered
#pragma unroll
      for (int r = 0; r < 8; ++r) {
        float val = fmaxf(acc[r] * gscale[oc0 + r] + gbias[oc0 + r], 0.0f);
        g_out[(((size_t)b * 8 + ((oc0 + r) >> 5)) * 32 + ((oc0 + r) & 31)) * (size_t)NP + p] =
            f2bf(val);
      }
    } else {
      // concat buffer u[b][p][512]; xf_g occupies channels 256..511 - contiguous
#pragma unroll
      for (int r = 0; r < 8; ++r)
        pk.u[r] = f2bf(fmaxf(acc[r] * gscale[oc0 + r] + gbias[oc0 + r], 0.0f));
      *(v8bf*)&g_out[((size_t)b * NP + p) * 512 + 256 + oc0] = pk.v;
    }
  }
}

__global__ __launch_bounds__(256)
void proj_kernel(const float* __restrict__ X, const u16* __restrict__ W,
                 const float* __restrict__ qbias,
                 const float* __restrict__ gscale, const float* __restrict__ gbias,
                 u16* __restrict__ qk_out, u16* __restrict__ g_out,
                 int NP, int is_zf) {
  __shared__ u16 lA[128][PKC + 8];
  __shared__ u16 lB[64][PKC + 8];

  const int b = blockIdx.z;
  const int m_base = blockIdx.x * 128;
  const int n_base = blockIdx.y * 64;
  const int t = threadIdx.x;
  const int wave = t >> 5;
  const int wm = wave >> 1;   // 0..3 : 32-row strip
  const int wn = wave & 1;    // 0..1 : 32-col strip
  const int K = 256;

  const float* Xb = X + (size_t)b * 256 * NP;

  v8f acc00 = {}, acc01 = {}, acc10 = {}, acc11 = {};

  for (int k0 = 0; k0 < K; k0 += PKC) {
    // Stage W tile 128 x PKC: b128 copies (bf16 row-major, 16B aligned)
#pragma unroll
    for (int c = t; c < (128 * PKC) / 8; c += 256) {
      int r = c >> 4;                 // PKC/8 = 16 chunks per row
      int kk = (c & 15) * 8;
      const u16* src = &W[(size_t)(m_base + r) * K + k0 + kk];
      *(v8bf*)&lA[r][kk] = *(const v8bf*)src;
      if (k0 + PKC < K) __builtin_prefetch(src + PKC, 0, 0);   // next chunk
    }
    // Stage X tile transposed to [n][k]: float4 along pixels, cvt to bf16
#pragma unroll
    for (int c = t; c < (64 * PKC) / 4; c += 256) {
      int kk = c >> 4;                // 16 pixel-quads per k-row
      int n4 = (c & 15) * 4;
      const float4 xv = *(const float4*)&Xb[(size_t)(k0 + kk) * NP + n_base + n4];
      lB[n4 + 0][kk] = f2bf(xv.x);
      lB[n4 + 1][kk] = f2bf(xv.y);
      lB[n4 + 2][kk] = f2bf(xv.z);
      lB[n4 + 3][kk] = f2bf(xv.w);
    }
    __syncthreads();
#pragma unroll
    for (int kk = 0; kk < PKC; kk += 32) {
      v16bf a0 = ld_frag(&lA[wm * 32][0],      PKC + 8, kk);
      v16bf a1 = ld_frag(&lA[wm * 32 + 16][0], PKC + 8, kk);
      v16bf b0 = ld_frag(&lB[wn * 32][0],      PKC + 8, kk);
      v16bf b1 = ld_frag(&lB[wn * 32 + 16][0], PKC + 8, kk);
      acc00 = WMMA_BF16(a0, b0, acc00);
      acc01 = WMMA_BF16(a0, b1, acc01);
      acc10 = WMMA_BF16(a1, b0, acc10);
      acc11 = WMMA_BF16(a1, b1, acc11);
    }
    __syncthreads();
  }

  // Epilogue. C layout: lane = n%16 + 16*(m>=8), vgpr r = m%8 within a tile,
  // so each lane holds 8 consecutive output channels at one pixel -> b128.
  const int lane = t & 31;
  const int nl = lane & 15;
  const int mh = (lane >> 4) * 8;
  const int ob0 = m_base + wm * 32 + mh;
  const int p0  = n_base + wn * 32 + nl;
  proj_store(acc00, ob0,      p0,      b, NP, is_zf, qbias, gscale, gbias, qk_out, g_out);
  proj_store(acc01, ob0,      p0 + 16, b, NP, is_zf, qbias, gscale, gbias, qk_out, g_out);
  proj_store(acc10, ob0 + 16, p0,      b, NP, is_zf, qbias, gscale, gbias, qk_out, g_out);
  proj_store(acc11, ob0 + 16, p0 + 16, b, NP, is_zf, qbias, gscale, gbias, qk_out, g_out);
}

// ---------------------------------------------------------------------------
// Attention per (batch, head): S = Q.K^T (K=32 -> one WMMA per 16x16 tile),
// in-register softmax over Nz=256, emb = P.V via LDS-staged P (bf16).
// Block: 128 threads = 4 waves; each wave owns 16 query rows.
// ---------------------------------------------------------------------------
__global__ __launch_bounds__(128)
void attn_kernel(const u16* __restrict__ qb, const u16* __restrict__ kb,
                 const u16* __restrict__ vb, u16* __restrict__ u_out) {
  __shared__ u16 lP[64][264];
  __shared__ u16 lV[32][264];

  const int b = blockIdx.z;
  const int h = blockIdx.y;
  const int nx0 = blockIdx.x * 64;
  const int t = threadIdx.x;
  const int wave = t >> 5;
  const int lane = t & 31;
  const size_t bh = (size_t)b * 8 + h;

  // Stage V: [dh=32][Nz=256], b128 copies
#pragma unroll
  for (int c = t; c < (32 * 256) / 8; c += 128) {
    int d = c >> 5;                  // 32 chunks per row
    int nz = (c & 31) * 8;
    *(v8bf*)&lV[d][nz] = *(const v8bf*)&vb[(bh * 32 + d) * 256 + nz];
  }

  // Q fragment for this wave's 16 rows (direct from global, layout matches)
  Frag fq;
  {
    const u16* qrow = qb + ((bh * 1024) + nx0 + wave * 16 + (lane & 15)) * 32 +
                      ((lane & 16) ? 8 : 0);
    fq.h[0] = *(const v8bf*)(qrow);
    fq.h[1] = *(const v8bf*)(qrow + 16);
  }

  // S = Q.K^T : 16 column tiles of 16, single K=32 WMMA each
  v8f S[16];
#pragma unroll
  for (int j = 0; j < 16; ++j) {
    Frag fk;
    const u16* krow = kb + ((bh * 256) + j * 16 + (lane & 15)) * 32 +
                      ((lane & 16) ? 8 : 0);
    fk.h[0] = *(const v8bf*)(krow);
    fk.h[1] = *(const v8bf*)(krow + 16);
    v8f z = {};
    S[j] = WMMA_BF16(fq.v, fk.v, z);
  }

  // Softmax over the 256 columns of each row. Row m (vgpr r) lives on one
  // 16-lane group -> shfl_xor reductions with masks 1,2,4,8 stay in-group.
#pragma unroll
  for (int r = 0; r < 8; ++r) {
    float m = S[0][r];
#pragma unroll
    for (int j = 1; j < 16; ++j) m = fmaxf(m, S[j][r]);
    m = fmaxf(m, __shfl_xor(m, 1, 32));
    m = fmaxf(m, __shfl_xor(m, 2, 32));
    m = fmaxf(m, __shfl_xor(m, 4, 32));
    m = fmaxf(m, __shfl_xor(m, 8, 32));
    float s = 0.0f;
#pragma unroll
    for (int j = 0; j < 16; ++j) {
      float e = __expf(S[j][r] - m);
      S[j][r] = e;
      s += e;
    }
    s += __shfl_xor(s, 1, 32);
    s += __shfl_xor(s, 2, 32);
    s += __shfl_xor(s, 4, 32);
    s += __shfl_xor(s, 8, 32);
    float inv = 1.0f / s;
    int mrow = wave * 16 + (lane >> 4) * 8 + r;
#pragma unroll
    for (int j = 0; j < 16; ++j)
      lP[mrow][j * 16 + (lane & 15)] = f2bf(S[j][r] * inv);
  }
  __syncthreads();   // covers lV staging + lP writes

  // emb = P @ V : M=16 (per wave), N=32 (two tiles), K=256
  v8f e0 = {}, e1 = {};
#pragma unroll
  for (int k0 = 0; k0 < 256; k0 += 32) {
    v16bf a  = ld_frag(&lP[wave * 16][0], 264, k0);
    v16bf b0 = ld_frag(&lV[0][0],  264, k0);
    v16bf b1 = ld_frag(&lV[16][0], 264, k0);
    e0 = WMMA_BF16(a, b0, e0);
    e1 = WMMA_BF16(a, b1, e1);
  }

  // emb channels 0..255 of concat buffer u[b][p][512]
  const int nl = lane & 15;
  const int mh = (lane >> 4) * 8;
#pragma unroll
  for (int r = 0; r < 8; ++r) {
    int p = nx0 + wave * 16 + mh + r;
    size_t rowbase = ((size_t)b * 1024 + p) * 512 + h * 32;
    u_out[rowbase + nl]      = f2bf(e0[r]);
    u_out[rowbase + 16 + nl] = f2bf(e1[r]);
  }
}

// ---------------------------------------------------------------------------
// Final conv: out[b,o,p] = relu(bn(fi_w[256,512] @ u[b][p][512])), f32 out.
// Same tiling as proj: 128x64 tile, K=512 in 4 chunks.
// ---------------------------------------------------------------------------
__global__ __launch_bounds__(256)
void fi_kernel(const u16* __restrict__ U, const u16* __restrict__ W,
               const float* __restrict__ fs, const float* __restrict__ fb,
               float* __restrict__ out) {
  __shared__ u16 lA[128][PKC + 8];
  __shared__ u16 lB[64][PKC + 8];

  const int b = blockIdx.z;
  const int m_base = blockIdx.x * 128;
  const int n_base = blockIdx.y * 64;
  const int t = threadIdx.x;
  const int wave = t >> 5;
  const int wm = wave >> 1;
  const int wn = wave & 1;
  const int K = 512;

  v8f acc00 = {}, acc01 = {}, acc10 = {}, acc11 = {};

  for (int k0 = 0; k0 < K; k0 += PKC) {
    // W tile: b128 copies
#pragma unroll
    for (int c = t; c < (128 * PKC) / 8; c += 256) {
      int r = c >> 4;
      int kk = (c & 15) * 8;
      const u16* src = &W[(size_t)(m_base + r) * K + k0 + kk];
      *(v8bf*)&lA[r][kk] = *(const v8bf*)src;
      if (k0 + PKC < K) __builtin_prefetch(src + PKC, 0, 0);
    }
    // U tile: already [p][c2] bf16 -> straight b128 copies
#pragma unroll
    for (int c = t; c < (64 * PKC) / 8; c += 256) {
      int n = c >> 4;
      int kk = (c & 15) * 8;
      const u16* src = &U[((size_t)b * 1024 + n_base + n) * 512 + k0 + kk];
      *(v8bf*)&lB[n][kk] = *(const v8bf*)src;
      if (k0 + PKC < K) __builtin_prefetch(src + PKC, 0, 0);
    }
    __syncthreads();
#pragma unroll
    for (int kk = 0; kk < PKC; kk += 32) {
      v16bf a0 = ld_frag(&lA[wm * 32][0],      PKC + 8, kk);
      v16bf a1 = ld_frag(&lA[wm * 32 + 16][0], PKC + 8, kk);
      v16bf b0 = ld_frag(&lB[wn * 32][0],      PKC + 8, kk);
      v16bf b1 = ld_frag(&lB[wn * 32 + 16][0], PKC + 8, kk);
      acc00 = WMMA_BF16(a0, b0, acc00);
      acc01 = WMMA_BF16(a0, b1, acc01);
      acc10 = WMMA_BF16(a1, b0, acc10);
      acc11 = WMMA_BF16(a1, b1, acc11);
    }
    __syncthreads();
  }

  const int lane = t & 31;
  const int nl = lane & 15;
  const int mh = (lane >> 4) * 8;
  const int ob0 = m_base + wm * 32 + mh;
  const int p0  = n_base + wn * 32 + nl;
#pragma unroll
  for (int r = 0; r < 8; ++r) {
    int o;
    o = ob0 + r;
    out[((size_t)b * 256 + o) * 1024 + p0]      = fmaxf(acc00[r] * fs[o] + fb[o], 0.0f);
    out[((size_t)b * 256 + o) * 1024 + p0 + 16] = fmaxf(acc01[r] * fs[o] + fb[o], 0.0f);
    o = ob0 + 16 + r;
    out[((size_t)b * 256 + o) * 1024 + p0]      = fmaxf(acc10[r] * fs[o] + fb[o], 0.0f);
    out[((size_t)b * 256 + o) * 1024 + p0 + 16] = fmaxf(acc11[r] * fs[o] + fb[o], 0.0f);
  }
}

// ---------------------------------------------------------------------------
// Host launcher
// ---------------------------------------------------------------------------
extern "C" void kernel_launch(void* const* d_in, const int* in_sizes, int n_in,
                              void* d_out, int out_size, void* d_ws, size_t ws_size,
                              hipStream_t stream) {
  const float* zf        = (const float*)d_in[0];
  const float* xf        = (const float*)d_in[1];
  const float* query_w   = (const float*)d_in[2];
  const float* query_b   = (const float*)d_in[3];
  const float* support_w = (const float*)d_in[4];
  const float* support_b = (const float*)d_in[5];
  const float* g_w       = (const float*)d_in[6];
  const float* g_b       = (const float*)d_in[7];
  const float* g_gamma   = (const float*)d_in[8];
  const float* g_beta    = (const float*)d_in[9];
  const float* g_mean    = (const float*)d_in[10];
  const float* g_var     = (const float*)d_in[11];
  const float* fi_w      = (const float*)d_in[12];
  const float* fi_b      = (const float*)d_in[13];
  const float* fi_gamma  = (const float*)d_in[14];
  const float* fi_beta   = (const float*)d_in[15];
  const float* fi_mean   = (const float*)d_in[16];
  const float* fi_var    = (const float*)d_in[17];
  float* out = (float*)d_out;

  char* ws = (char*)d_ws;
  size_t off = 0;
  auto alloc = [&](size_t bytes) -> void* {
    void* p = ws + off;
    off = (off + bytes + 255) & ~(size_t)255;
    return p;
  };

  u16*   wqg  = (u16*)alloc(512 * 256 * sizeof(u16));      // [query_w; g_w] bf16
  u16*   wsg  = (u16*)alloc(512 * 256 * sizeof(u16));      // [support_w; g_w] bf16
  u16*   wfi  = (u16*)alloc(256 * 512 * sizeof(u16));      // fi_w bf16
  float* gs   = (float*)alloc(256 * sizeof(float));
  float* gb2  = (float*)alloc(256 * sizeof(float));
  float* fs   = (float*)alloc(256 * sizeof(float));
  float* fb2  = (float*)alloc(256 * sizeof(float));
  u16*   qbuf = (u16*)alloc((size_t)64 * 8 * 1024 * 32 * sizeof(u16)); // q [b][h][nx][d]
  u16*   kbuf = (u16*)alloc((size_t)64 * 8 * 256 * 32 * sizeof(u16));  // k [b][h][nz][d]
  u16*   vbuf = (u16*)alloc((size_t)64 * 8 * 32 * 256 * sizeof(u16));  // v [b][h][d][nz]
  u16*   ubuf = (u16*)alloc((size_t)64 * 1024 * 512 * sizeof(u16));    // concat [b][p][2C]

  // 1. weight conversion + BN folding
  {
    const int total = 2 * 256 * 256 + 256 * 512 + 512;
    prep_kernel<<<(total + 255) / 256, 256, 0, stream>>>(
        query_w, support_w, g_w, fi_w,
        g_b, g_gamma, g_beta, g_mean, g_var,
        fi_b, fi_gamma, fi_beta, fi_mean, fi_var,
        wqg, wsg, wfi, gs, gb2, fs, fb2);
  }
  // 2. xf projections: q + xf_g   (M=512, N=1024, per batch)
  proj_kernel<<<dim3(4, 16, 64), 256, 0, stream>>>(
      xf, wqg, query_b, gs, gb2, qbuf, ubuf, 1024, /*is_zf=*/0);
  // 3. zf projections: k + v      (M=512, N=256, per batch)
  proj_kernel<<<dim3(4, 4, 64), 256, 0, stream>>>(
      zf, wsg, support_b, gs, gb2, kbuf, vbuf, 256, /*is_zf=*/1);
  // 4. attention + emb -> channels 0..255 of concat buffer
  attn_kernel<<<dim3(16, 8, 64), 128, 0, stream>>>(qbuf, kbuf, vbuf, ubuf);
  // 5. final fi conv + BN + ReLU -> f32 output [B,256,32,32]
  fi_kernel<<<dim3(2, 16, 64), 256, 0, stream>>>(ubuf, wfi, fs, fb2, out);
}
